// BidirectionalAttention_57698590654997
// MI455X (gfx1250) — compile-verified, matrix operand order checked by reading, not run on hardware
//
#include <hip/hip_runtime.h>
#include <hip/hip_bf16.h>
#include <math.h>

// ---- problem constants ----
#define BB   2
#define NSEQ 2048
#define DIM  1024
#define HH   16
#define DH   64
#define M_ROWS (BB*NSEQ)        // 4096
#define NQKV  (3*DIM)           // 3072

typedef __bf16 bf16;
typedef __attribute__((ext_vector_type(16))) __bf16 v16bf;
typedef __attribute__((ext_vector_type(8)))  __bf16 v8bf;
typedef __attribute__((ext_vector_type(8)))  float  v8f;

// ---------------- WMMA helpers ----------------
__device__ __forceinline__ v8f wmma_bf16(v16bf a, v16bf b, v8f c) {
  // D = A(16x32 bf16) * B(32x16 bf16) + C(16x16 f32)
  return __builtin_amdgcn_wmma_f32_16x16x32_bf16(
      /*neg_a=*/false, a, /*neg_b=*/false, b,
      /*c_mod=*/(short)0, c, /*reuse_a=*/false, /*reuse_b=*/false);
}

// Load a 16-element bf16 fragment slice for one lane from a K-contiguous row.
// A-frag: rowptr = &A[(m0 + (lane&15)) * ld],  B-frag: rowptr = &BT[(n0 + (lane&15)) * ld]
// element e -> k = k0 + e + 8*((e>>3) + half), realized as two 16-byte loads.
__device__ __forceinline__ v16bf load_frag(const bf16* rowptr, int k0, int half) {
  const v8bf lo = *(const v8bf*)(rowptr + k0 + 8 * half);
  const v8bf hi = *(const v8bf*)(rowptr + k0 + 16 + 8 * half);
  v16bf a;
#pragma unroll
  for (int i = 0; i < 8; ++i) { a[i] = lo[i]; a[i + 8] = hi[i]; }
  return a;
}

#define V8F_ZERO ((v8f){0.f,0.f,0.f,0.f,0.f,0.f,0.f,0.f})

__device__ __forceinline__ void mac8(v8f acc[2][4], v16bf a0, v16bf a1,
                                     v16bf b0, v16bf b1, v16bf b2, v16bf b3) {
  acc[0][0] = wmma_bf16(a0, b0, acc[0][0]);
  acc[0][1] = wmma_bf16(a0, b1, acc[0][1]);
  acc[0][2] = wmma_bf16(a0, b2, acc[0][2]);
  acc[0][3] = wmma_bf16(a0, b3, acc[0][3]);
  acc[1][0] = wmma_bf16(a1, b0, acc[1][0]);
  acc[1][1] = wmma_bf16(a1, b1, acc[1][1]);
  acc[1][2] = wmma_bf16(a1, b2, acc[1][2]);
  acc[1][3] = wmma_bf16(a1, b3, acc[1][3]);
}

// 32(M) x 64(N) x DIM(K) wave-level GEMM core.
// Manual ping-pong double buffering (k step 64 = 2 wmma k-steps): buffers are
// always refilled by loads, never copied, so no register-rotation movs at the
// loop backedge.
__device__ __forceinline__ void gemm_core_32x64(
    const bf16* arow0, const bf16* arow1,
    const bf16* brow0, const bf16* brow1, const bf16* brow2, const bf16* brow3,
    int half, v8f acc[2][4]) {
  v16bf A0 = load_frag(arow0, 0, half);
  v16bf A1 = load_frag(arow1, 0, half);
  v16bf B0 = load_frag(brow0, 0, half);
  v16bf B1 = load_frag(brow1, 0, half);
  v16bf B2 = load_frag(brow2, 0, half);
  v16bf B3 = load_frag(brow3, 0, half);

  int k0 = 0;
  for (; k0 < DIM - 64; k0 += 64) {
    const v16bf A0b = load_frag(arow0, k0 + 32, half);
    const v16bf A1b = load_frag(arow1, k0 + 32, half);
    const v16bf B0b = load_frag(brow0, k0 + 32, half);
    const v16bf B1b = load_frag(brow1, k0 + 32, half);
    const v16bf B2b = load_frag(brow2, k0 + 32, half);
    const v16bf B3b = load_frag(brow3, k0 + 32, half);

    mac8(acc, A0, A1, B0, B1, B2, B3);

    A0 = load_frag(arow0, k0 + 64, half);
    A1 = load_frag(arow1, k0 + 64, half);
    B0 = load_frag(brow0, k0 + 64, half);
    B1 = load_frag(brow1, k0 + 64, half);
    B2 = load_frag(brow2, k0 + 64, half);
    B3 = load_frag(brow3, k0 + 64, half);

    mac8(acc, A0b, A1b, B0b, B1b, B2b, B3b);
  }
  // tail: k0 == DIM-64; 'A/B' hold the k0 fragments
  {
    const v16bf A0b = load_frag(arow0, k0 + 32, half);
    const v16bf A1b = load_frag(arow1, k0 + 32, half);
    const v16bf B0b = load_frag(brow0, k0 + 32, half);
    const v16bf B1b = load_frag(brow1, k0 + 32, half);
    const v16bf B2b = load_frag(brow2, k0 + 32, half);
    const v16bf B3b = load_frag(brow3, k0 + 32, half);
    mac8(acc, A0, A1, B0, B1, B2, B3);
    mac8(acc, A0b, A1b, B0b, B1b, B2b, B3b);
  }
}

// ---------------- 1) RMSNorm -> bf16 ----------------
__global__ __launch_bounds__(256)
void rmsnorm_kernel(const float* __restrict__ x, const float* __restrict__ g,
                    bf16* __restrict__ xn) {
  __shared__ float red[256];
  const int row = blockIdx.x;            // 0..4095
  const int tid = threadIdx.x;           // 0..255, 4 elems each
  const size_t base = (size_t)row * DIM + tid * 4;
  float v0 = x[base + 0], v1 = x[base + 1], v2 = x[base + 2], v3 = x[base + 3];
  float ss = v0 * v0 + v1 * v1 + v2 * v2 + v3 * v3;
  red[tid] = ss;
  __syncthreads();
#pragma unroll
  for (int s = 128; s > 0; s >>= 1) {
    if (tid < s) red[tid] += red[tid + s];
    __syncthreads();
  }
  const float norm = sqrtf(red[0]);
  const float inv  = 32.0f / fmaxf(norm, 1e-12f);   // sqrt(DIM)=32
  const int c = tid * 4;
  xn[base + 0] = (bf16)(v0 * inv * g[c + 0]);
  xn[base + 1] = (bf16)(v1 * inv * g[c + 1]);
  xn[base + 2] = (bf16)(v2 * inv * g[c + 2]);
  xn[base + 3] = (bf16)(v3 * inv * g[c + 3]);
}

// ---------------- 2) transpose f32 weight -> bf16, K-contiguous ----------------
__global__ __launch_bounds__(256)
void transpose_w_kernel(const float* __restrict__ in, bf16* __restrict__ out,
                        int rows, int cols) {
  const size_t idx = (size_t)blockIdx.x * 256 + threadIdx.x;
  const size_t r = idx / cols, c = idx % cols;
  out[c * (size_t)rows + r] = (bf16)in[idx];
}

// ---------------- 3) QKV GEMM: xn[4096,1024] @ w_qkv -> q,k raw + vT ----------------
__global__ __launch_bounds__(256)
void gemm_qkv_kernel(const bf16* __restrict__ xn, const bf16* __restrict__ wT,
                     bf16* __restrict__ q, bf16* __restrict__ k, bf16* __restrict__ vT) {
  const int wave = threadIdx.x >> 5, lane = threadIdx.x & 31;
  const int half = lane >> 4, ln = lane & 15;
  const int tile = blockIdx.x * 8 + wave;          // over (M/32)*(NQKV/64)
  const int nt_n = NQKV / 64;                      // 48
  const int m0 = (tile / nt_n) * 32;
  const int n0 = (tile % nt_n) * 64;

  v8f acc[2][4];
#pragma unroll
  for (int gm = 0; gm < 2; ++gm)
#pragma unroll
    for (int t = 0; t < 4; ++t) acc[gm][t] = V8F_ZERO;

  gemm_core_32x64(xn + (size_t)(m0 + ln) * DIM,
                  xn + (size_t)(m0 + 16 + ln) * DIM,
                  wT + (size_t)(n0 + ln) * DIM,
                  wT + (size_t)(n0 + 16 + ln) * DIM,
                  wT + (size_t)(n0 + 32 + ln) * DIM,
                  wT + (size_t)(n0 + 48 + ln) * DIM,
                  half, acc);

  // epilogue: route to q/k [B,H,N,DH] and vT [B,H,DH,N]
#pragma unroll
  for (int gm = 0; gm < 2; ++gm) {
#pragma unroll
    for (int t = 0; t < 4; ++t) {
#pragma unroll
      for (int r = 0; r < 8; ++r) {
        const float val = acc[gm][t][r];
        const int row = m0 + gm * 16 + r + 8 * half;   // 0..4095
        const int col = n0 + t * 16 + ln;              // 0..3071
        const int sec = col >> 10;                     // 0=q 1=k 2=v
        const int cc  = col & 1023;
        const int h   = cc >> 6, d = cc & 63;
        const int b   = row >> 11, nn = row & 2047;
        const size_t bh = (size_t)b * HH + h;
        if (sec == 0)      q[(bh * NSEQ + nn) * DH + d] = (bf16)val;
        else if (sec == 1) k[(bh * NSEQ + nn) * DH + d] = (bf16)val;
        else               vT[(bh * DH + d) * NSEQ + nn] = (bf16)val;
      }
    }
  }
}

// ---------------- 4) RoPE (DH=64, partner = idx^32) ----------------
__global__ __launch_bounds__(256)
void rope_kernel(const bf16* __restrict__ src, const float* __restrict__ pos,
                 bf16* __restrict__ dst) {
  const size_t idx = (size_t)blockIdx.x * 256 + threadIdx.x;  // over B*H*N*DH
  const int d  = (int)(idx & 63);
  const int nn = (int)((idx >> 6) & (NSEQ - 1));
  const float pe = pos[nn * DH + d];
  const float c = cosf(pe), s = sinf(pe);
  const float v  = (float)src[idx];
  const float vp = (float)src[idx ^ 32];
  const float sign = (d < 32) ? -1.0f : 1.0f;
  dst[idx] = (bf16)(v * c + sign * vp * s);
}

// ---------------- 5) Flash attention (per wave: 16 query rows x DH=64) ----------------
__global__ __launch_bounds__(128)
void attn_kernel(const bf16* __restrict__ q, const bf16* __restrict__ k,
                 const bf16* __restrict__ vT, bf16* __restrict__ aout) {
  __shared__ bf16 plds[4][16][32];                 // per-wave P tile staging
  const int wave = threadIdx.x >> 5, lane = threadIdx.x & 31;
  const int half = lane >> 4, ln = lane & 15;
  const int mt = blockIdx.x & 31;                  // N/64 = 32 row-tiles
  const int bh = blockIdx.x >> 5;                  // b*16 + h
  const size_t bh_off = (size_t)bh * NSEQ * DH;
  const int m0 = mt * 64 + wave * 16;

  const bf16* qrow  = q + bh_off + (size_t)(m0 + ln) * DH;
  const bf16* kbase = k + bh_off;
  const v16bf qa0 = load_frag(qrow, 0, half);
  const v16bf qa1 = load_frag(qrow, 32, half);

  const bf16* vrow[4];
#pragma unroll
  for (int t = 0; t < 4; ++t) vrow[t] = vT + bh_off + (size_t)(t * 16 + ln) * NSEQ;

  v8f o[4];
#pragma unroll
  for (int t = 0; t < 4; ++t) o[t] = V8F_ZERO;
  float mrun[8], lrun[8];
#pragma unroll
  for (int r = 0; r < 8; ++r) { mrun[r] = -1e30f; lrun[r] = 0.f; }

  // prologue: K fragments for j0 = 0
  v16bf kf00 = load_frag(kbase + (size_t)(0 + ln) * DH, 0, half);
  v16bf kf01 = load_frag(kbase + (size_t)(0 + ln) * DH, 32, half);
  v16bf kf10 = load_frag(kbase + (size_t)(16 + ln) * DH, 0, half);
  v16bf kf11 = load_frag(kbase + (size_t)(16 + ln) * DH, 32, half);

  for (int j0 = 0; j0 < NSEQ; j0 += 32) {
    // S = Q K^T for two 16-wide key subtiles (K-dim 64 = 2 wmma steps)
    v8f c0 = wmma_bf16(qa0, kf00, V8F_ZERO);
    c0 = wmma_bf16(qa1, kf01, c0);
    v8f c1 = wmma_bf16(qa0, kf10, V8F_ZERO);
    c1 = wmma_bf16(qa1, kf11, c1);

    // V-tile loads for this j0: overlap the softmax math
    const v16bf vb0 = load_frag(vrow[0], j0, half);
    const v16bf vb1 = load_frag(vrow[1], j0, half);
    const v16bf vb2 = load_frag(vrow[2], j0, half);
    const v16bf vb3 = load_frag(vrow[3], j0, half);

    // prefetch next iteration's K fragments (wrapped index keeps the last
    // iteration branchless and in-bounds; the wrapped data is never consumed)
    const int jn = (j0 + 32) & (NSEQ - 1);
    kf00 = load_frag(kbase + (size_t)(jn + ln) * DH, 0, half);
    kf01 = load_frag(kbase + (size_t)(jn + ln) * DH, 32, half);
    kf10 = load_frag(kbase + (size_t)(jn + 16 + ln) * DH, 0, half);
    kf11 = load_frag(kbase + (size_t)(jn + 16 + ln) * DH, 32, half);

    float s0[8], s1[8], p0[8], p1[8];
#pragma unroll
    for (int r = 0; r < 8; ++r) { s0[r] = c0[r] * 0.125f; s1[r] = c1[r] * 0.125f; }

    // online softmax: rows live across 16 lanes of a half-wave
#pragma unroll
    for (int r = 0; r < 8; ++r) {
      float mloc = fmaxf(s0[r], s1[r]);
#pragma unroll
      for (int msk = 8; msk >= 1; msk >>= 1)
        mloc = fmaxf(mloc, __shfl_xor(mloc, msk, 32));
      const float mnew  = fmaxf(mrun[r], mloc);
      const float alpha = __expf(mrun[r] - mnew);
      p0[r] = __expf(s0[r] - mnew);
      p1[r] = __expf(s1[r] - mnew);
      float ps = p0[r] + p1[r];
#pragma unroll
      for (int msk = 8; msk >= 1; msk >>= 1)
        ps += __shfl_xor(ps, msk, 32);
      lrun[r] = lrun[r] * alpha + ps;
      mrun[r] = mnew;
#pragma unroll
      for (int t = 0; t < 4; ++t) o[t][r] *= alpha;
    }

    // restage P (C layout) -> A-fragment layout via LDS
#pragma unroll
    for (int r = 0; r < 8; ++r) {
      const int row = r + 8 * half;
      plds[wave][row][ln]      = (bf16)p0[r];
      plds[wave][row][16 + ln] = (bf16)p1[r];
    }
    __syncthreads();
    const v16bf pa = load_frag(&plds[wave][ln][0], 0, half);
    __syncthreads();

    // O += P(16x32) @ V(32x64), B-frags from transposed V
    o[0] = wmma_bf16(pa, vb0, o[0]);
    o[1] = wmma_bf16(pa, vb1, o[1]);
    o[2] = wmma_bf16(pa, vb2, o[2]);
    o[3] = wmma_bf16(pa, vb3, o[3]);
  }

  // finalize + store to attn_out [B*N, DIM] as bf16
  const int b = bh >> 4, h = bh & 15;
#pragma unroll
  for (int r = 0; r < 8; ++r) {
    const float inv = 1.0f / lrun[r];
    const size_t row = (size_t)b * NSEQ + m0 + r + 8 * half;
#pragma unroll
    for (int t = 0; t < 4; ++t)
      aout[row * DIM + h * DH + t * 16 + ln] = (bf16)(o[t][r] * inv);
  }
}

// ---------------- 6) out projection: aout[4096,1024] @ w_out -> f32 ----------------
__global__ __launch_bounds__(256)
void gemm_out_kernel(const bf16* __restrict__ a_in, const bf16* __restrict__ wT,
                     float* __restrict__ out) {
  const int wave = threadIdx.x >> 5, lane = threadIdx.x & 31;
  const int half = lane >> 4, ln = lane & 15;
  const int tile = blockIdx.x * 8 + wave;          // over (M/32)*(DIM/64)
  const int nt_n = DIM / 64;                       // 16
  const int m0 = (tile / nt_n) * 32;
  const int n0 = (tile % nt_n) * 64;

  v8f acc[2][4];
#pragma unroll
  for (int gm = 0; gm < 2; ++gm)
#pragma unroll
    for (int t = 0; t < 4; ++t) acc[gm][t] = V8F_ZERO;

  gemm_core_32x64(a_in + (size_t)(m0 + ln) * DIM,
                  a_in + (size_t)(m0 + 16 + ln) * DIM,
                  wT + (size_t)(n0 + ln) * DIM,
                  wT + (size_t)(n0 + 16 + ln) * DIM,
                  wT + (size_t)(n0 + 32 + ln) * DIM,
                  wT + (size_t)(n0 + 48 + ln) * DIM,
                  half, acc);

#pragma unroll
  for (int gm = 0; gm < 2; ++gm)
#pragma unroll
    for (int t = 0; t < 4; ++t)
#pragma unroll
      for (int r = 0; r < 8; ++r)
        out[(size_t)(m0 + gm * 16 + r + 8 * half) * DIM + n0 + t * 16 + ln] = acc[gm][t][r];
}

// ---------------- launch ----------------
extern "C" void kernel_launch(void* const* d_in, const int* in_sizes, int n_in,
                              void* d_out, int out_size, void* d_ws, size_t ws_size,
                              hipStream_t stream) {
  const float* x     = (const float*)d_in[0];   // [B,N,DIM]
  const float* pos   = (const float*)d_in[1];   // [N,DH]
  const float* g     = (const float*)d_in[2];   // [DIM]
  const float* w_qkv = (const float*)d_in[3];   // [DIM, 3*DIM]
  const float* w_out = (const float*)d_in[4];   // [DIM, DIM]
  float* out = (float*)d_out;

  char* ws = (char*)d_ws;
  const size_t SZ_BHND = (size_t)BB * HH * NSEQ * DH * sizeof(bf16);  // 8 MB
  bf16* xn     = (bf16*)(ws);                                   // 8 MB
  bf16* wqkvT  = (bf16*)(ws + 8388608);                         // 6 MB
  bf16* woutT  = (bf16*)(ws + 8388608 + 6291456);               // 2 MB
  bf16* qraw   = (bf16*)(ws + 16777216);
  bf16* kraw   = (bf16*)(ws + 16777216 + SZ_BHND);
  bf16* qrope  = (bf16*)(ws + 16777216 + 2 * SZ_BHND);
  bf16* krope  = (bf16*)(ws + 16777216 + 3 * SZ_BHND);
  bf16* vT     = (bf16*)(ws + 16777216 + 4 * SZ_BHND);
  bf16* aout   = (bf16*)(ws + 16777216 + 5 * SZ_BHND);          // total 64 MB

  rmsnorm_kernel<<<M_ROWS, 256, 0, stream>>>(x, g, xn);
  transpose_w_kernel<<<(DIM * NQKV) / 256, 256, 0, stream>>>(w_qkv, wqkvT, DIM, NQKV);
  transpose_w_kernel<<<(DIM * DIM) / 256, 256, 0, stream>>>(w_out, woutT, DIM, DIM);
  gemm_qkv_kernel<<<(M_ROWS / 32) * (NQKV / 64) / 8, 256, 0, stream>>>(xn, wqkvT, qraw, kraw, vT);
  rope_kernel<<<(BB * HH * NSEQ * DH) / 256, 256, 0, stream>>>(qraw, pos, qrope);
  rope_kernel<<<(BB * HH * NSEQ * DH) / 256, 256, 0, stream>>>(kraw, pos, krope);
  attn_kernel<<<BB * HH * (NSEQ / 64), 128, 0, stream>>>(qrope, krope, vT, aout);
  gemm_out_kernel<<<(M_ROWS / 32) * (DIM / 64) / 8, 256, 0, stream>>>(aout, woutT, out);
}
